// NonlinearFilter_51436528337752
// MI455X (gfx1250) — compile-verified, matrix operand order checked by reading, not run on hardware
//
#include <hip/hip_runtime.h>

// ---------------------------------------------------------------------------
// MI455X (gfx1250) nonlinear phase filter:
//   phi = conv1d(|x|^2, Nkernel)  via v_wmma_f32_16x16x32_f16 (Toeplitz-in-A
//   mapping, 100% N-utilization), then fused theta-rotation epilogue.
// ---------------------------------------------------------------------------

typedef __attribute__((ext_vector_type(16))) _Float16 v16h;
typedef __attribute__((ext_vector_type(8)))  float    v8f;

constexpr int   NB     = 16;        // batches
constexpr int   MFULL  = 200000;    // samples per batch per mode
constexpr int   MOUT   = 199600;    // valid conv outputs (MFULL - 400)
constexpr int   HHALF  = 200;       // NTAPS/2
constexpr int   NCHUNK = 17;        // ceil(513/32): tap-chunk sweep
constexpr int   MP     = 200704;    // padded power-plane length (covers halo)
constexpr float THFAC  = 0.0016567f * 2000000.0f / 25.0f; // GAMMA*LFIB/STEP

constexpr size_t A_HALFS = (size_t)NCHUNK * 2 * 32 * 16;  // A fragments
constexpr size_t A_BYTES = A_HALFS * 2;                   // 34816 B

union Frag {
    unsigned int u[8];
    uint4        u4[2];
    v16h         h;
};

// ---------------------------------------------------------------------------
// Kernel 1: precompute WMMA A-fragments from Nkernel.
// Fragment (c,i): A[r,k] = Nk[o][i][32c + K - 16j], r = o*8+j, laid out in the
// exact per-lane 16-bit A layout so the main loop loads 2x b128 per lane.
// grid = 34 (c*2+i), block = 32 lanes.
// ---------------------------------------------------------------------------
__global__ void nlf_prep_afrag(const float* __restrict__ nk,
                               _Float16* __restrict__ afr)
{
    const int c    = blockIdx.x >> 1;
    const int i    = blockIdx.x & 1;
    const int lane = threadIdx.x;        // 0..31
    const int r    = lane & 15;
    const int o    = r >> 3;
    const int j    = r & 7;
    const int kb   = (lane < 16) ? 0 : 8;

    _Float16* dst = afr + ((size_t)((c * 2 + i) * 32 + lane)) * 16;
    #pragma unroll
    for (int q = 0; q < 2; ++q) {
        #pragma unroll
        for (int e = 0; e < 8; ++e) {
            const int K = kb + q * 16 + e;
            const int t = 32 * c + K - 16 * j;
            float v = 0.0f;
            if (t >= 0 && t <= 400) v = nk[(o * 2 + i) * 401 + t];
            dst[q * 8 + e] = (_Float16)v;
        }
    }
}

// ---------------------------------------------------------------------------
// Kernel 2: power = xr^2 + xi^2 -> f16, stored twice per (b,mode):
//   plane s=0: p[x],  plane s=1: p[x] written at index x-1 (shift-by-one copy)
// so every B-fragment lane load is 4-byte aligned (parity trick).
// Pad region [MFULL, MP) zero-filled.
// ---------------------------------------------------------------------------
__global__ void nlf_prep_power(const float* __restrict__ xr,
                               const float* __restrict__ xi,
                               _Float16* __restrict__ pp)
{
    const int x = blockIdx.x * blockDim.x + threadIdx.x;
    const int b = blockIdx.y;
    if (x >= MP) return;

    float p0 = 0.0f, p1 = 0.0f;
    if (x < MFULL) {
        const size_t base = ((size_t)b * MFULL + x) * 2;
        const float r0 = xr[base], r1 = xr[base + 1];
        const float i0 = xi[base], i1 = xi[base + 1];
        p0 = r0 * r0 + i0 * i0;
        p1 = r1 * r1 + i1 * i1;
    }
    const _Float16 h0 = (_Float16)p0;
    const _Float16 h1 = (_Float16)p1;

    const size_t pl00 = (size_t)(b * 4 + 0) * MP;   // i=0, s=0
    const size_t pl01 = (size_t)(b * 4 + 1) * MP;   // i=0, s=1
    const size_t pl10 = (size_t)(b * 4 + 2) * MP;   // i=1, s=0
    const size_t pl11 = (size_t)(b * 4 + 3) * MP;   // i=1, s=1

    pp[pl00 + x] = h0;
    pp[pl10 + x] = h1;
    if (x > 0) {
        pp[pl01 + x - 1] = h0;
        pp[pl11 + x - 1] = h1;
    } else {
        pp[pl01 + MP - 1] = (_Float16)0.0f;
        pp[pl11 + MP - 1] = (_Float16)0.0f;
    }
}

// ---------------------------------------------------------------------------
// Kernel 3: main WMMA conv + rotation.
// One wave -> one 16x16 f32 D tile = phi for 128 consecutive outputs x 2 modes.
//   D[r=o*8+j, n] = sum_{c,i,k} Nk[o,i,32c+k-16j] * power[b, m0+32c+k+n, i]
// Block = 128 threads (4 waves, 512 outputs). grid = (ceil(MOUT/512), 16).
// ---------------------------------------------------------------------------
__global__ __launch_bounds__(128, 1)
void nlf_main(const float* __restrict__ xr, const float* __restrict__ xi,
              const float* __restrict__ tinfo,
              const _Float16* __restrict__ afr,
              const _Float16* __restrict__ pp,
              float* __restrict__ out)
{
    const int b    = blockIdx.y;
    const int lane = threadIdx.x & 31;
    const int wave = threadIdx.x >> 5;
    const int m0   = blockIdx.x * 512 + wave * 128;
    const int n    = lane & 15;
    const int kb   = (lane & 16) >> 1;   // 0 (lanes 0-15) or 8 (lanes 16-31)
    const int s    = n & 1;              // parity plane select

    v8f acc = {0.f, 0.f, 0.f, 0.f, 0.f, 0.f, 0.f, 0.f};

    const uint4* __restrict__ a4 = (const uint4*)afr;

    for (int c = 0; c < NCHUNK; ++c) {
        #pragma unroll
        for (int i = 0; i < 2; ++i) {
            Frag A, Bf;
            // A fragment: per-lane 32 B, 16-byte aligned, L2-resident
            const uint4* ap = a4 + ((size_t)((c * 2 + i) * 32 + lane)) * 2;
            A.u4[0] = ap[0];
            A.u4[1] = ap[1];

            // B fragment: shifted window from the parity plane matching n&1.
            // Half-index is even by construction -> aligned b32 loads.
            const size_t pbase = (size_t)((b * 2 + i) * 2 + s) * MP
                               + (size_t)(m0 + 32 * c + kb + (n - s));
            const unsigned int* pu = (const unsigned int*)(pp + pbase);
            Bf.u[0] = pu[0];  Bf.u[1] = pu[1];  Bf.u[2] = pu[2];  Bf.u[3] = pu[3];
            Bf.u[4] = pu[8];  Bf.u[5] = pu[9];  Bf.u[6] = pu[10]; Bf.u[7] = pu[11];
            __builtin_prefetch((const void*)(pu + 16), 0, 0);

            acc = __builtin_amdgcn_wmma_f32_16x16x32_f16(
                      false, A.h, false, Bf.h, (short)0, acc, false, false);
        }
    }

    // Epilogue: theta = phi * (GAMMA*LFIB/STEP) * P[b];  rotate cropped x.
    const float ti    = tinfo[b * 4];
    const float scale = THFAC * 0.0005f * exp2f(ti * 0.3321928095f); // 10^(ti/10)
    const int   o     = lane >> 4;   // output mode of this half-wave

    #pragma unroll
    for (int v = 0; v < 8; ++v) {
        const int m = m0 + 16 * v + n;   // D row M = v + 8*o, col = n
        if (m < MOUT) {
            const float th = acc[v] * scale;
            float sn, cs;
            __sincosf(th, &sn, &cs);
            const size_t xidx = ((size_t)b * MFULL + (m + HHALF)) * 2 + o;
            const float xrv = xr[xidx];
            const float xiv = xi[xidx];
            float2 r2;
            r2.x = xrv * cs - xiv * sn;
            r2.y = xrv * sn + xiv * cs;
            *(float2*)(out + (((size_t)b * MOUT + m) * 2 + o) * 2) = r2;
        }
    }
}

// ---------------------------------------------------------------------------
extern "C" void kernel_launch(void* const* d_in, const int* in_sizes, int n_in,
                              void* d_out, int out_size, void* d_ws, size_t ws_size,
                              hipStream_t stream)
{
    (void)in_sizes; (void)n_in; (void)out_size; (void)ws_size;

    const float* xr = (const float*)d_in[0];   // (16, 200000, 2) f32
    const float* xi = (const float*)d_in[1];   // (16, 200000, 2) f32
    const float* ti = (const float*)d_in[2];   // (16, 4) f32
    const float* nk = (const float*)d_in[3];   // (2, 2, 401) f32
    float* out = (float*)d_out;                // (16, 199600, 2, 2) f32

    _Float16* afr = (_Float16*)d_ws;                           // 34816 B
    _Float16* pp  = (_Float16*)((char*)d_ws + A_BYTES);        // ~25.7 MB

    nlf_prep_afrag<<<dim3(NCHUNK * 2), dim3(32), 0, stream>>>(nk, afr);

    nlf_prep_power<<<dim3((MP + 255) / 256, NB), dim3(256), 0, stream>>>(xr, xi, pp);

    const int blocks_x = (MOUT + 511) / 512;   // 390
    nlf_main<<<dim3(blocks_x, NB), dim3(128), 0, stream>>>(xr, xi, ti, afr, pp, out);
}
